// AttentiveModel_8615704395880
// MI455X (gfx1250) — matrix operand order repaired and checked by reading, not compiled, not run on hardware
//
#include <hip/hip_runtime.h>
#include <hip/hip_bf16.h>

// ---------------- types / helpers ----------------
typedef __attribute__((ext_vector_type(16))) __bf16 v16bf;
typedef __attribute__((ext_vector_type(8)))  float  v8f;

__device__ __forceinline__ unsigned short f2bf(float x) {
    unsigned u = __float_as_uint(x);
    unsigned r = u + 0x7FFFu + ((u >> 16) & 1u);   // round-to-nearest-even
    return (unsigned short)(r >> 16);
}
__device__ __forceinline__ float bf2f(unsigned short h) {
    return __uint_as_float(((unsigned)h) << 16);
}

union FragU { uint4 q[2]; v16bf v; };

// ---------------- setup kernels ----------------
__global__ void conv_f32_bf16(const float* __restrict__ in, unsigned short* __restrict__ out, int n) {
    int i = blockIdx.x * blockDim.x + threadIdx.x;
    if (i < n) out[i] = f2bf(in[i]);
}

__global__ void mask_kernel(const float* __restrict__ enc, unsigned char* __restrict__ mask, int BN) {
    int i = blockIdx.x * blockDim.x + threadIdx.x;
    if (i < BN) mask[i] = (enc[(size_t)i * 256] == 0.0f) ? 1 : 0;
}

// Wcat[256,256] = [Wd(128) | W_log(78) | zero pad(50)], bcat likewise
__global__ void wcat_kernel(const float* __restrict__ Wd, const float* __restrict__ Wlog,
                            const float* __restrict__ bd, const float* __restrict__ blog,
                            unsigned short* __restrict__ Wcat, float* __restrict__ bcat) {
    int i = blockIdx.x * blockDim.x + threadIdx.x;
    if (i < 256 * 256) {
        int k = i >> 8, c = i & 255;
        float v = (c < 128) ? Wd[k * 128 + c] : ((c < 206) ? Wlog[k * 78 + (c - 128)] : 0.0f);
        Wcat[i] = f2bf(v);
    }
    if (i < 256) bcat[i] = (i < 128) ? bd[i] : ((i < 206) ? blog[i - 128] : 0.0f);
}

// ---------------- generic bf16 WMMA GEMM ----------------
// C[M,N] = A[M,K] (bf16 rowmajor, optional row gather) @ Bw[K,N] (bf16 rowmajor) + bias
// Requirements: M % 16 == 0, N % 64 == 0, K % 32 == 0.
// Each wave computes a 16(M) x 64(N) tile: 4 accumulators, A-frag reused across 4 WMMAs.
__global__ void __launch_bounds__(128)
gemm_bf16_wmma(const unsigned short* __restrict__ A, int lda,
               const int* __restrict__ rowidx, int idx_stride, int idx_off,
               const unsigned short* __restrict__ Bw, int ldb,
               const float* __restrict__ bias,
               float* __restrict__ Cf, unsigned short* __restrict__ Cb, int ldc,
               int M, int N, int K) {
    const int lane = threadIdx.x & 31;
    const int wave = blockIdx.x * (blockDim.x >> 5) + (threadIdx.x >> 5);
    const int ntiles64 = N >> 6;
    const int mt = wave / ntiles64;
    const int nt = wave % ntiles64;
    if (mt * 16 >= M) return;
    const int m0 = mt * 16;
    const int n0 = nt * 64;

    // A fragment addressing (ISA 7.12.2, 16-bit A 16x32):
    // lanes 0-15: row=lane, K {0..7,16..23}; lanes 16-31: row=lane-16, K {8..15,24..31}
    const int arow = m0 + (lane & 15);
    const unsigned short* aptr =
        rowidx ? (A + (size_t)rowidx[arow * idx_stride + idx_off] * lda)
               : (A + (size_t)arow * lda);
    const int kadd = (lane >> 4) ? 8 : 0;

    v8f zero = {0.f, 0.f, 0.f, 0.f, 0.f, 0.f, 0.f, 0.f};
    v8f acc0 = zero, acc1 = zero, acc2 = zero, acc3 = zero;

    for (int k0 = 0; k0 < K; k0 += 32) {
        FragU a;
        a.q[0] = *reinterpret_cast<const uint4*>(aptr + k0 + kadd);
        a.q[1] = *reinterpret_cast<const uint4*>(aptr + k0 + kadd + 16);
        // B fragment: lane holds col N=n0+(lane&15), K in same order as A (kadd + {0..7,16..23})
        const unsigned short* bcolp = Bw + n0 + (lane & 15);
        FragU b0, b1, b2, b3;
#pragma unroll
        for (int s = 0; s < 16; s++) {
            const unsigned short* br = bcolp + (size_t)(k0 + kadd + (s < 8 ? s : s + 8)) * ldb;
            reinterpret_cast<unsigned short*>(&b0)[s] = br[0];
            reinterpret_cast<unsigned short*>(&b1)[s] = br[16];
            reinterpret_cast<unsigned short*>(&b2)[s] = br[32];
            reinterpret_cast<unsigned short*>(&b3)[s] = br[48];
        }
        acc0 = __builtin_amdgcn_wmma_f32_16x16x32_bf16(false, a.v, false, b0.v, (short)0, acc0, false, false);
        acc1 = __builtin_amdgcn_wmma_f32_16x16x32_bf16(false, a.v, false, b1.v, (short)0, acc1, false, false);
        acc2 = __builtin_amdgcn_wmma_f32_16x16x32_bf16(false, a.v, false, b2.v, (short)0, acc2, false, false);
        acc3 = __builtin_amdgcn_wmma_f32_16x16x32_bf16(false, a.v, false, b3.v, (short)0, acc3, false, false);
    }

    // C/D layout: elem e of lane l -> row m0 + e + (l>=16 ? 8:0), col n0 + (l&15)
    const int crow = m0 + ((lane >> 4) ? 8 : 0);
    const int ccol = n0 + (lane & 15);
    v8f accs[4] = {acc0, acc1, acc2, acc3};
    float bv[4];
#pragma unroll
    for (int j = 0; j < 4; j++) bv[j] = bias ? bias[ccol + j * 16] : 0.0f;

    if (Cf) {
#pragma unroll
        for (int j = 0; j < 4; j++)
#pragma unroll
            for (int e = 0; e < 8; e++)
                Cf[(size_t)(crow + e) * ldc + ccol + j * 16] = accs[j][e] + bv[j];
    }
    if (Cb) {
#pragma unroll
        for (int j = 0; j < 4; j++)
#pragma unroll
            for (int e = 0; e < 8; e++)
                Cb[(size_t)(crow + e) * ldc + ccol + j * 16] = f2bf(accs[j][e] + bv[j]);
    }
}

// ---------------- GRU elementwise ----------------
__global__ void gru_kernel(const float* __restrict__ gi, const float* __restrict__ gh,
                           const float* __restrict__ hprev,
                           float* __restrict__ hnew_f, unsigned short* __restrict__ hnew_b) {
    int i = blockIdx.x * blockDim.x + threadIdx.x;
    if (i >= 256 * 256) return;
    int b = i >> 8, j = i & 255;
    const float* gib = gi + (size_t)b * 768;
    const float* ghb = gh + (size_t)b * 768;
    float r = 1.0f / (1.0f + __expf(-(gib[j] + ghb[j])));
    float z = 1.0f / (1.0f + __expf(-(gib[256 + j] + ghb[256 + j])));
    float n = tanhf(gib[512 + j] + r * ghb[512 + j]);
    float h = hprev[i];
    float hn = (1.0f - z) * n + z * h;
    hnew_f[i] = hn;
    hnew_b[i] = f2bf(hn);
}

// ---------------- attention (one block per batch row) ----------------
// scores: tanh(enc_e[b,n,:]+dproj[b,:]) . Wo + bo, mask, softmax over N=512,
// attn[b,:] = sum_n p[n]*enc[b,n,:]; state = base + attn; copy logits to out.
__global__ void __launch_bounds__(256)
attention_kernel(const unsigned short* __restrict__ enc_bf,   // [B,512,256]
                 const unsigned short* __restrict__ ence_bf,  // [B,512,128]
                 const unsigned char* __restrict__ mask,      // [B,512]
                 const float* __restrict__ P,                 // [B,256]
                 const float* __restrict__ Wo, const float* __restrict__ bo,
                 const float* __restrict__ base,              // [B,256]
                 float* __restrict__ state_f, unsigned short* __restrict__ state_b,
                 float* __restrict__ out, int t) {
    __shared__ float s_dp[128];
    __shared__ float s_wo[128];
    __shared__ float s_p[512];
    __shared__ float s_red[256];
    const int b = blockIdx.x;
    const int tid = threadIdx.x;
    if (tid < 128) { s_dp[tid] = P[(size_t)b * 256 + tid]; s_wo[tid] = Wo[tid]; }
    __syncthreads();

    float sc[2];
#pragma unroll
    for (int h = 0; h < 2; h++) {
        int n = tid + h * 256;
        float s;
        if (!mask[(size_t)b * 512 + n]) {
            s = -1e9f;
        } else {
            const unsigned int* row =
                (const unsigned int*)(ence_bf + ((size_t)b * 512 + n) * 128);
            float acc = 0.0f;
#pragma unroll 4
            for (int k2 = 0; k2 < 64; k2++) {
                unsigned int w = row[k2];
                float f0 = __uint_as_float(w << 16);
                float f1 = __uint_as_float(w & 0xFFFF0000u);
                int k = k2 * 2;
                acc += tanhf(f0 + s_dp[k]) * s_wo[k];
                acc += tanhf(f1 + s_dp[k + 1]) * s_wo[k + 1];
            }
            s = acc + bo[0];
        }
        sc[h] = s;
    }
    // block max
    s_red[tid] = fmaxf(sc[0], sc[1]);
    __syncthreads();
    for (int off = 128; off > 0; off >>= 1) {
        if (tid < off) s_red[tid] = fmaxf(s_red[tid], s_red[tid + off]);
        __syncthreads();
    }
    float mx = s_red[0];
    __syncthreads();
    // exp + sum
    float e0 = __expf(sc[0] - mx), e1 = __expf(sc[1] - mx);
    s_p[tid] = e0; s_p[tid + 256] = e1;
    s_red[tid] = e0 + e1;
    __syncthreads();
    for (int off = 128; off > 0; off >>= 1) {
        if (tid < off) s_red[tid] += s_red[tid + off];
        __syncthreads();
    }
    float inv = 1.0f / s_red[0];
    s_p[tid] *= inv; s_p[tid + 256] *= inv;
    __syncthreads();

    // weighted sum over enc (coalesced: lanes read contiguous e at fixed n)
    const unsigned short* col = enc_bf + (size_t)b * 512 * 256 + tid;
    float acc = 0.0f;
    for (int n = 0; n < 512; n++) acc += s_p[n] * bf2f(col[(size_t)n * 256]);
    float st = base[(size_t)b * 256 + tid] + acc;
    state_f[(size_t)b * 256 + tid] = st;
    state_b[(size_t)b * 256 + tid] = f2bf(st);

    if (t >= 0 && tid < 78)
        out[((size_t)b * 64 + t) * 78 + tid] = P[(size_t)b * 256 + 128 + tid];
}

// ---------------- host launcher ----------------
extern "C" void kernel_launch(void* const* d_in, const int* in_sizes, int n_in,
                              void* d_out, int out_size, void* d_ws, size_t ws_size,
                              hipStream_t stream) {
    const int B = 256, N = 512, T = 64, ENC = 256, EMB = 128, ATTN = 128, V = 78;
    (void)in_sizes; (void)n_in; (void)out_size; (void)ws_size;

    const float* h_0   = (const float*)d_in[0];
    const float* enc   = (const float*)d_in[1];
    const int*   toks  = (const int*)  d_in[2];
    const float* embT  = (const float*)d_in[3];
    const float* W_ds  = (const float*)d_in[4];
    const float* b_ds  = (const float*)d_in[5];
    const float* We    = (const float*)d_in[6];
    const float* be    = (const float*)d_in[7];
    const float* Wd    = (const float*)d_in[8];
    const float* bd    = (const float*)d_in[9];
    const float* Wo    = (const float*)d_in[10];
    const float* bo    = (const float*)d_in[11];
    const float* W_ih  = (const float*)d_in[12];
    const float* b_ih  = (const float*)d_in[13];
    const float* W_hh  = (const float*)d_in[14];
    const float* b_hh  = (const float*)d_in[15];
    const float* W_log = (const float*)d_in[16];
    const float* b_log = (const float*)d_in[17];
    float* out = (float*)d_out;

    // bump allocator over d_ws (256B aligned)
    char* wsbase = (char*)d_ws;
    size_t off = 0;
    auto alloc = [&](size_t bytes) -> char* {
        char* p = wsbase + off;
        off = (off + bytes + 255) & ~(size_t)255;
        return p;
    };
    unsigned short* enc_b  = (unsigned short*)alloc((size_t)B * N * ENC * 2);   // 67 MB
    unsigned short* ence_b = (unsigned short*)alloc((size_t)B * N * ATTN * 2);  // 33.5 MB
    unsigned char*  maskb  = (unsigned char*)alloc((size_t)B * N);
    unsigned short* We_b   = (unsigned short*)alloc((size_t)ENC * ATTN * 2);
    unsigned short* Wcat_b = (unsigned short*)alloc((size_t)ENC * 256 * 2);
    float*          bcat_f = (float*)alloc(256 * 4);
    unsigned short* Wih_b  = (unsigned short*)alloc((size_t)EMB * 768 * 2);
    unsigned short* Whh_b  = (unsigned short*)alloc((size_t)ENC * 768 * 2);
    unsigned short* Wds_b  = (unsigned short*)alloc((size_t)ENC * ENC * 2);
    unsigned short* emb_b  = (unsigned short*)alloc((size_t)V * EMB * 2);
    unsigned short* h0_b   = (unsigned short*)alloc((size_t)B * ENC * 2);
    float*          dec_f  = (float*)alloc((size_t)B * ENC * 4);
    unsigned short* dec_b  = (unsigned short*)alloc((size_t)B * ENC * 2);
    float*          state_f= (float*)alloc((size_t)B * ENC * 4);
    unsigned short* state_b= (unsigned short*)alloc((size_t)B * ENC * 2);
    float*          gi     = (float*)alloc((size_t)B * 768 * 4);
    float*          gh     = (float*)alloc((size_t)B * 768 * 4);
    float*          hnew_f = (float*)alloc((size_t)B * ENC * 4);
    unsigned short* hnew_b = (unsigned short*)alloc((size_t)B * ENC * 2);
    float*          P      = (float*)alloc((size_t)B * 256 * 4);

    auto conv = [&](const float* src, unsigned short* dst, int n) {
        conv_f32_bf16<<<(n + 255) / 256, 256, 0, stream>>>(src, dst, n);
    };
    // ---- setup: conversions ----
    conv(enc,   enc_b, B * N * ENC);
    conv(We,    We_b,  ENC * ATTN);
    conv(W_ih,  Wih_b, EMB * 768);
    conv(W_hh,  Whh_b, ENC * 768);
    conv(W_ds,  Wds_b, ENC * ENC);
    conv(embT,  emb_b, V * EMB);
    conv(h_0,   h0_b,  B * ENC);
    mask_kernel<<<(B * N + 255) / 256, 256, 0, stream>>>(enc, maskb, B * N);
    wcat_kernel<<<(256 * 256 + 255) / 256, 256, 0, stream>>>(Wd, W_log, bd, b_log, Wcat_b, bcat_f);

    auto gemm = [&](const unsigned short* A, int lda, const int* ridx, int istr, int ioff,
                    const unsigned short* Bw, int ldb, const float* bias,
                    float* Cf, unsigned short* Cb, int ldc, int M, int Nn, int K) {
        int ntiles64 = Nn >> 6;                       // Nn % 64 == 0
        int waves = (M / 16) * ntiles64;
        int blocks = (waves + 3) / 4;                 // 4 waves per 128-thread block
        gemm_bf16_wmma<<<blocks, 128, 0, stream>>>(A, lda, ridx, istr, ioff,
                                                   Bw, ldb, bias, Cf, Cb, ldc, M, Nn, K);
    };

    // enc_e = enc @ We + be  (bf16 output, L2-resident for all 64 steps)
    gemm(enc_b, ENC, nullptr, 0, 0, We_b, ATTN, be, nullptr, ence_b, ATTN, B * N, ATTN, ENC);
    // dec_start = h_0 @ W_ds + b_ds
    gemm(h0_b, ENC, nullptr, 0, 0, Wds_b, ENC, b_ds, dec_f, dec_b, ENC, B, ENC, ENC);
    // P0 = dec_start @ Wcat + bcat  (dproj in cols 0..127, logits 128..205)
    gemm(dec_b, ENC, nullptr, 0, 0, Wcat_b, 256, bcat_f, P, nullptr, 256, B, 256, ENC);
    // state0 = dec_start + attn(dec_start)
    attention_kernel<<<B, 256, 0, stream>>>(enc_b, ence_b, maskb, P, Wo, bo,
                                            dec_f, state_f, state_b, out, -1);

    // ---- decode loop ----
    for (int t = 0; t < T; t++) {
        // gi = emb_table[tokens[:,t]] @ W_ih + b_ih   (row-gathered A)
        gemm(emb_b, EMB, toks, T, t, Wih_b, 768, b_ih, gi, nullptr, 768, B, 768, EMB);
        // gh = h @ W_hh + b_hh
        gemm(state_b, ENC, nullptr, 0, 0, Whh_b, 768, b_hh, gh, nullptr, 768, B, 768, ENC);
        // GRU gates -> hnew
        gru_kernel<<<(B * ENC + 255) / 256, 256, 0, stream>>>(gi, gh, state_f, hnew_f, hnew_b);
        // P = hnew @ [Wd | W_log | pad] + [bd | b_log | 0]
        gemm(hnew_b, ENC, nullptr, 0, 0, Wcat_b, 256, bcat_f, P, nullptr, 256, B, 256, ENC);
        // attention + state update + logits copy to out[:, t, :]
        attention_kernel<<<B, 256, 0, stream>>>(enc_b, ence_b, maskb, P, Wo, bo,
                                                hnew_f, state_f, state_b, out, t);
    }
}
